// GraphAttention_43224550868108
// MI455X (gfx1250) — matrix operand order repaired and checked by reading, not compiled, not run on hardware
//
#include <hip/hip_runtime.h>

typedef __attribute__((ext_vector_type(16))) _Float16 v16h;
typedef __attribute__((ext_vector_type(8)))  float    v8f;

#define NNODES 8192
#define INDIM  256
#define HIDDEN 64
#define HEADS  2
#define L2E    1.4426950408889634f

// ---------------------------------------------------------------------------
// Kernel 1: feat = X @ W[h]  (f32), store featT (f16, [head][c][node]),
//           s = feat @ a_self, n = feat @ a_neigh   (f32)
// Block: 64 threads, handles 8 nodes for one head. W[h] read once per block.
// ---------------------------------------------------------------------------
__global__ __launch_bounds__(64)
void gat_prep(const float* __restrict__ X, const float* __restrict__ W,
              const float* __restrict__ a_self, const float* __restrict__ a_neigh,
              _Float16* __restrict__ featT, float* __restrict__ sv,
              float* __restrict__ nv)
{
    __shared__ float xs[8][INDIM];     // 8 KB
    __shared__ float red[2][64];

    const int tid  = threadIdx.x;
    const int head = blockIdx.y;
    const int i0   = blockIdx.x * 8;

    // stage 8 rows of X (coalesced float4)
    const float4* X4  = (const float4*)(X + (size_t)i0 * INDIM);
    float4*       xs4 = (float4*)&xs[0][0];
#pragma unroll
    for (int q = 0; q < 8; ++q) xs4[tid + 64 * q] = X4[tid + 64 * q];
    __syncthreads();

    float acc[8] = {0.f,0.f,0.f,0.f,0.f,0.f,0.f,0.f};
    const float* Wp = W + (size_t)head * INDIM * HIDDEN + tid;   // column tid
    for (int k = 0; k < INDIM; ++k) {
        const float w = Wp[(size_t)k * HIDDEN];
#pragma unroll
        for (int r = 0; r < 8; ++r) acc[r] = fmaf(xs[r][k], w, acc[r]);
    }

    const float as = a_self [head * HIDDEN + tid];
    const float an = a_neigh[head * HIDDEN + tid];

#pragma unroll 1
    for (int r = 0; r < 8; ++r) {
        featT[(size_t)head * HIDDEN * NNODES + (size_t)tid * NNODES + i0 + r] =
            (_Float16)acc[r];
        red[0][tid] = acc[r] * as;
        red[1][tid] = acc[r] * an;
#pragma unroll
        for (int off = 32; off >= 1; off >>= 1) {
            __syncthreads();
            if (tid < off) {
                red[0][tid] += red[0][tid + off];
                red[1][tid] += red[1][tid + off];
            }
        }
        if (tid == 0) {
            sv[head * NNODES + i0 + r] = red[0][0];
            nv[head * NNODES + i0 + r] = red[1][0];
        }
        __syncthreads();
    }
}

// ---------------------------------------------------------------------------
// Kernel 2: fused attention. One workgroup (4 waves) per (16-row block, head).
// Each wave handles j-chunks kt = wv, wv+4, ... ; per chunk builds the 16x32
// f16 weight matrix directly in WMMA A-operand layout and accumulates
// 4x v_wmma_f32_16x16x32_f16 into 16x64 f32.
// ---------------------------------------------------------------------------
__global__ __launch_bounds__(128)
void gat_attn(const float* __restrict__ A, const _Float16* __restrict__ featT,
              const float* __restrict__ sv, const float* __restrict__ nv,
              const float* __restrict__ bias, float* __restrict__ out)
{
    __shared__ float Atile[4][16 * 36];        // per-wave A stage, pad 36 (bank-safe, 16B aligned)
    __shared__ float numred[4][16][HIDDEN];    // 16 KB cross-wave partials
    __shared__ float denred[4][16];

    const int tid  = threadIdx.x;
    const int wv   = tid >> 5;
    const int lane = tid & 31;
    const int head = blockIdx.x & 1;           // interleave heads -> L2 reuse of A rows
    const int i0   = (blockIdx.x >> 1) * 16;

    const int r     = lane & 15;               // A-operand row this lane owns
    const int kbase = (lane & 16) ? 16 : 0;    // contiguous K order (k = 16h + s): hi half-wave K+16
    const int jhalf = (lane & 16) ? 16 : 0;    // contiguous K order, same as A: hi half-wave K+16
    const int ncol  = lane & 15;               // B-operand column this lane owns

    const float     s_r = sv[head * NNODES + i0 + r];
    const float*    nvh = nv + head * NNODES;
    const _Float16* fT  = featT + (size_t)head * HIDDEN * NNODES;

    v8f acc0 = {0.f,0.f,0.f,0.f,0.f,0.f,0.f,0.f};
    v8f acc1 = acc0, acc2 = acc0, acc3 = acc0;
    float den = 0.f;

    for (int kt = wv; kt < NNODES / 32; kt += 4) {
        const int j0 = kt * 32;

        // ---- stage A tile 16x32 f32 (coalesced b128; same-wave LDS is in-order)
#pragma unroll
        for (int q = 0; q < 4; ++q) {
            const int f = lane + 32 * q;
            const int row = f >> 3, c4 = f & 7;
            const float4 vA =
                *(const float4*)(A + (size_t)(i0 + row) * NNODES + j0 + c4 * 4);
            *(float4*)&Atile[wv][row * 36 + c4 * 4] = vA;
        }
        // prefetch next tile for this wave (global_prefetch_b8)
        if (kt + 4 < NNODES / 32)
            __builtin_prefetch(A + (size_t)(i0 + (lane >> 1)) * NNODES +
                                   (kt + 4) * 32 + (lane & 1) * 16, 0, 3);

        // ---- gather the 16 n_j values this lane needs (8 aligned float2)
        float2 npair[8];
#pragma unroll
        for (int p = 0; p < 8; ++p) {
            const int k = kbase + 2 * p;
            npair[p] = *(const float2*)(nvh + j0 + k);
        }

        // ---- weights: w = exp2(L2E * exp2(-L2E*d^2)) * A   (== exp(exp(-d^2))*A)
        v16h aop;
#pragma unroll
        for (int p = 0; p < 8; ++p) {
            const int k = kbase + 2 * p;
            const float2 av = *(const float2*)&Atile[wv][r * 36 + k];
            const float d0 = s_r - npair[p].x;
            const float d1 = s_r - npair[p].y;
            const float w0 =
                __builtin_amdgcn_exp2f(L2E * __builtin_amdgcn_exp2f(-L2E * d0 * d0)) * av.x;
            const float w1 =
                __builtin_amdgcn_exp2f(L2E * __builtin_amdgcn_exp2f(-L2E * d1 * d1)) * av.y;
            den += w0 + w1;
            aop[2 * p]     = (_Float16)w0;
            aop[2 * p + 1] = (_Float16)w1;
        }

        // ---- B operands: featT[c][j0+jhalf .. +15] is exactly one lane's v16h
        const _Float16* fbase = fT + j0 + jhalf;
        const v16h b0 = *(const v16h*)(fbase + (size_t)(ncol +  0) * NNODES);
        const v16h b1 = *(const v16h*)(fbase + (size_t)(ncol + 16) * NNODES);
        const v16h b2 = *(const v16h*)(fbase + (size_t)(ncol + 32) * NNODES);
        const v16h b3 = *(const v16h*)(fbase + (size_t)(ncol + 48) * NNODES);

        acc0 = __builtin_amdgcn_wmma_f32_16x16x32_f16(false, aop, false, b0,
                                                      (short)0, acc0, false, false);
        acc1 = __builtin_amdgcn_wmma_f32_16x16x32_f16(false, aop, false, b1,
                                                      (short)0, acc1, false, false);
        acc2 = __builtin_amdgcn_wmma_f32_16x16x32_f16(false, aop, false, b2,
                                                      (short)0, acc2, false, false);
        acc3 = __builtin_amdgcn_wmma_f32_16x16x32_f16(false, aop, false, b3,
                                                      (short)0, acc3, false, false);
    }

    // ---- fold half-wave denominators: lanes l and l^16 cover complementary K sets
    den += __shfl_xor(den, 16, 32);
    if (lane < 16) denred[wv][lane] = den;

    // ---- spill numerator accumulators (C/D layout: M = v + 8*(lane>>4), N = lane&15)
    const int m_off = (lane >> 4) << 3;
#pragma unroll
    for (int v = 0; v < 8; ++v) {
        numred[wv][v + m_off][ncol +  0] = acc0[v];
        numred[wv][v + m_off][ncol + 16] = acc1[v];
        numred[wv][v + m_off][ncol + 32] = acc2[v];
        numred[wv][v + m_off][ncol + 48] = acc3[v];
    }
    __syncthreads();

    // ---- cross-wave reduce, normalize, bias, relu, store
#pragma unroll
    for (int q = 0; q < 8; ++q) {
        const int idx = tid + 128 * q;
        const int m = idx >> 6, c = idx & 63;
        const float num = numred[0][m][c] + numred[1][m][c] +
                          numred[2][m][c] + numred[3][m][c];
        const float d4 = denred[0][m] + denred[1][m] + denred[2][m] + denred[3][m];
        const float val = num / d4 + bias[head * HIDDEN + c];
        out[(size_t)(i0 + m) * (HEADS * HIDDEN) + head * HIDDEN + c] =
            fmaxf(val, 0.f);
    }
}

// ---------------------------------------------------------------------------
extern "C" void kernel_launch(void* const* d_in, const int* in_sizes, int n_in,
                              void* d_out, int out_size, void* d_ws, size_t ws_size,
                              hipStream_t stream)
{
    const float* X       = (const float*)d_in[0];   // [8192,256]
    const float* A       = (const float*)d_in[1];   // [8192,8192]
    const float* W       = (const float*)d_in[2];   // [2,256,64]
    const float* b       = (const float*)d_in[3];   // [2,64]
    const float* a_self  = (const float*)d_in[4];   // [2,64]
    const float* a_neigh = (const float*)d_in[5];   // [2,64]
    float*       out     = (float*)d_out;           // [8192,128]

    char* ws = (char*)d_ws;
    _Float16* featT = (_Float16*)ws;                               // 2 MiB
    float*    sv    = (float*)(ws + (size_t)HEADS * HIDDEN * NNODES * sizeof(_Float16));
    float*    nv    = sv + HEADS * NNODES;

    gat_prep<<<dim3(NNODES / 8, HEADS), 64, 0, stream>>>(X, W, a_self, a_neigh,
                                                         featT, sv, nv);
    gat_attn<<<dim3((NNODES / 16) * HEADS), 128, 0, stream>>>(A, featT, sv, nv,
                                                              b, out);
}
